// CNNRNNModel_23295902613610
// MI455X (gfx1250) — compile-verified, matrix operand order checked by reading, not run on hardware
//
#include <hip/hip_runtime.h>

// ---------------------------------------------------------------------------
// CDNA5 (gfx1250) CNN(ResNet18-ish) + 4-layer LSTM decoder.
// - convs / fc / proj / LSTM gates: v_wmma_f32_16x16x32_bf16 implicit GEMM
// - weights pre-packed once into WMMA-B-fragment bf16 order
// - LSTM h/x state kept in bf16; staged whole into LDS with
//   global_load_async_to_lds_b128 (ASYNCcnt) -> barrier-free WMMA K-loop
// - LSTM packed weights (~59MB bf16) stay resident in 192MB L2 for 100 steps
// ---------------------------------------------------------------------------

typedef __attribute__((ext_vector_type(16))) __bf16 v16bf;
typedef __attribute__((ext_vector_type(8)))  __bf16 v8bf;
typedef __attribute__((ext_vector_type(8)))  float  v8f;

__device__ __forceinline__ __bf16 f2bf(float f) {
  unsigned u = __builtin_bit_cast(unsigned, f);
  unsigned r = u + 0x7FFFu + ((u >> 16) & 1u);      // round-to-nearest-even
  unsigned short h = (unsigned short)(r >> 16);
  return __builtin_bit_cast(__bf16, h);
}
__device__ __forceinline__ float bf2f(__bf16 b) {
  unsigned u = (unsigned)__builtin_bit_cast(unsigned short, b) << 16;
  return __builtin_bit_cast(float, u);
}

// A-fragment from an LDS tile with row stride `stride` (bf16 elems), K offset
// kbase. ISA 16-bit A layout: lanes 0-15 row=lane, K={0..7,16..23}; lanes
// 16-31 row=lane&15, K={8..15,24..31}. Two 16B runs -> two ds_load_b128.
__device__ __forceinline__ v16bf lds_a_frag(const __bf16* sA, int stride,
                                            int kbase, int lane) {
  int row = lane & 15;
  int khalf = (lane >> 4) << 3;
  const __bf16* p = sA + row * stride + kbase;
  v8bf lo = *(const v8bf*)(p + khalf);
  v8bf hi = *(const v8bf*)(p + 16 + khalf);
  return __builtin_shufflevector(lo, hi, 0,1,2,3,4,5,6,7,8,9,10,11,12,13,14,15);
}

// B-fragment: packed contiguous per (ntile,ktile): 32 lanes x 16 bf16.
__device__ __forceinline__ v16bf load_b_frag(const __bf16* __restrict__ Bpk,
                                             long tile, int lane) {
  const __bf16* p = Bpk + tile * 512 + lane * 16;   // 32B aligned
  v8bf lo = *(const v8bf*)(p);
  v8bf hi = *(const v8bf*)(p + 8);
  return __builtin_shufflevector(lo, hi, 0,1,2,3,4,5,6,7,8,9,10,11,12,13,14,15);
}

__device__ __forceinline__ v8f wmma_bf16(v16bf a, v16bf b, v8f c) {
  return __builtin_amdgcn_wmma_f32_16x16x32_bf16(false, a, false, b, (short)0, c,
                                                 false, false);
}

// CDNA5 async global->LDS copy (tracked by ASYNCcnt). lds byte offset is the
// low 32 bits of the flat LDS address (ISA flat-aperture mapping).
__device__ __forceinline__ void async_g2l_b128(void* lds_dst, const void* gsrc) {
  unsigned ldsa = (unsigned)(unsigned long long)lds_dst;
  asm volatile("global_load_async_to_lds_b128 %0, %1, off"
               :: "v"(ldsa), "v"(gsrc) : "memory");
}
__device__ __forceinline__ void wait_async0() {
  asm volatile("s_wait_asynccnt 0" ::: "memory");
}

__device__ __forceinline__ float sigf(float x) { return 1.f / (1.f + __expf(-x)); }

// ---------------------------------------------------------------------------
// Weight packer: src row-major [N][K] f32 -> bf16 fragment-ordered tiles.
// ---------------------------------------------------------------------------
__global__ void pack_b_kernel(const float* __restrict__ src, __bf16* __restrict__ dst,
                              int N, int K, int Ksteps, long total) {
  long i = (long)blockIdx.x * blockDim.x + threadIdx.x;
  if (i >= total) return;
  long tile = i >> 9;
  int rem = (int)(i & 511);
  int lane = rem >> 4, e = rem & 15;
  long ntile = tile / Ksteps;
  int kt = (int)(tile - ntile * Ksteps);
  int n = (int)(ntile << 4) + (lane & 15);
  int khalf = (lane >> 4) << 3;
  int k = (kt << 5) + ((e < 8) ? (khalf + e) : (16 + khalf + (e - 8)));
  float v = (n < N && k < K) ? src[(long)n * K + k] : 0.f;
  dst[i] = f2bf(v);
}

__global__ void zero_kernel(float* p, int n) {
  int i = blockIdx.x * blockDim.x + threadIdx.x;
  if (i < n) p[i] = 0.f;
}

// ---------------------------------------------------------------------------
// Implicit-GEMM conv, fused bias/BN(eval)/residual/ReLU.
// 128 thr (4 waves); blockIdx.x = m-tile (16 pixels), blockIdx.y*4+wave =
// n-tile (16 out channels). 16x64 K-slab staged per barrier -> 2 WMMAs.
// ---------------------------------------------------------------------------
__global__ void conv_wmma_kernel(
    const float* __restrict__ in, const __bf16* __restrict__ wpk,
    const float* __restrict__ bias,
    const float* __restrict__ bng, const float* __restrict__ bnb,
    const float* __restrict__ bnm, const float* __restrict__ bnv,
    const float* __restrict__ resid, float* __restrict__ out,
    int Bn, int Cin, int H, int W, int Cout, int Ho, int Wo,
    int KH, int KW, int stride, int pad, int do_relu) {
  __shared__ __align__(16) __bf16 sA[16 * 64];
  const int lane = threadIdx.x & 31;
  const int wave = threadIdx.x >> 5;
  const int mtile = blockIdx.x;
  const int ntile = blockIdx.y * 4 + wave;
  const int M = Bn * Ho * Wo;
  const int K = Cin * KH * KW;
  const int Ksteps = (K + 31) >> 5;
  const int Kpairs = (K + 63) >> 6;
  const int ntiles = (Cout + 15) >> 4;
  const int HoWo = Ho * Wo;
  const int KHKW = KH * KW;
  const bool nvalid = (ntile < ntiles);

  v8f acc = {};
  for (int kp = 0; kp < Kpairs; ++kp) {
    for (int i = threadIdx.x; i < 1024; i += 128) {
      int row = i >> 6, kl = i & 63;
      int m = (mtile << 4) + row;
      int k = (kp << 6) + kl;
      float v = 0.f;
      if (m < M && k < K) {
        int b = m / HoWo; int rem = m - b * HoWo;
        int oh = rem / Wo; int ow = rem - oh * Wo;
        int ci = k / KHKW; int r2 = k - ci * KHKW;
        int kh = r2 / KW;  int kw = r2 - kh * KW;
        int ih = oh * stride - pad + kh;
        int iw = ow * stride - pad + kw;
        if (ih >= 0 && ih < H && iw >= 0 && iw < W)
          v = in[((long)(b * Cin + ci) * H + ih) * W + iw];
      }
      sA[i] = f2bf(v);
    }
    __syncthreads();
    if (nvalid) {
      v16bf a0 = lds_a_frag(sA, 64, 0, lane);
      v16bf b0 = load_b_frag(wpk, (long)ntile * Ksteps + 2 * kp, lane);
      acc = wmma_bf16(a0, b0, acc);
      if (2 * kp + 1 < Ksteps) {
        v16bf a1 = lds_a_frag(sA, 64, 32, lane);
        v16bf b1 = load_b_frag(wpk, (long)ntile * Ksteps + 2 * kp + 1, lane);
        acc = wmma_bf16(a1, b1, acc);
      }
    }
    __syncthreads();
  }
  if (!nvalid) return;
  int co = (ntile << 4) + (lane & 15);
  if (co >= Cout) return;
  float inv = 1.f, shf = 0.f;
  if (bng) { inv = bng[co] * rsqrtf(bnv[co] + 1e-5f); shf = bnb[co] - bnm[co] * inv; }
  float bs = bias ? bias[co] : 0.f;
  for (int r = 0; r < 8; ++r) {
    int m = (mtile << 4) + r + ((lane >> 4) << 3);   // C/D: VGPR r -> M=r / 8+r
    if (m >= M) continue;
    float v = (acc[r] + bs) * inv + shf;
    int b = m / HoWo; int rem = m - b * HoWo;
    int oh = rem / Wo; int ow = rem - oh * Wo;
    long o = ((long)(b * Cout + co) * Ho + oh) * Wo + ow;
    if (resid) v += resid[o];
    if (do_relu) v = fmaxf(v, 0.f);
    out[o] = v;
  }
}

__global__ void maxpool_kernel(const float* __restrict__ in, float* __restrict__ out,
                               int Bn, int C, int H, int W, int Ho, int Wo) {
  long total = (long)Bn * C * Ho * Wo;
  long i = (long)blockIdx.x * blockDim.x + threadIdx.x;
  if (i >= total) return;
  int ow = (int)(i % Wo); long r = i / Wo;
  int oh = (int)(r % Ho); r /= Ho;
  int c = (int)(r % C);   int b = (int)(r / C);
  float m = -3.4e38f;
  for (int kh = 0; kh < 3; ++kh)
    for (int kw = 0; kw < 3; ++kw) {
      int ih = oh * 2 - 1 + kh, iw = ow * 2 - 1 + kw;
      if (ih >= 0 && ih < H && iw >= 0 && iw < W)
        m = fmaxf(m, in[((long)(b * C + c) * H + ih) * W + iw]);
    }
  out[i] = m;
}

// mean over HW -> padded 16 x C (rows >= Bn are zero)
__global__ void avgpool_kernel(const float* __restrict__ in, float* __restrict__ feat,
                               int Bn, int C, int HW) {
  int i = blockIdx.x * blockDim.x + threadIdx.x;
  if (i >= 16 * C) return;
  int c = i % C, row = i / C;
  float v = 0.f;
  if (row < Bn) {
    const float* p = in + (long)(row * C + c) * HW;
    for (int k = 0; k < HW; ++k) v += p[k];
    v /= (float)HW;
  }
  feat[row * C + c] = v;
}

// ---------------------------------------------------------------------------
// GEMM, M=16 (padded batch): out = A[16xK]*Bpk + bias.
// mode 0: f32 [16][N]; mode 1: f32 LSTM-split (n -> layer=n>>10, j=n&1023,
// dst[(layer*16+row)*1024+j]); mode 2: same split but bf16 output (h state).
// ---------------------------------------------------------------------------
__global__ void gemm16_wmma_kernel(const float* __restrict__ A, int K,
                                   const __bf16* __restrict__ Bpk,
                                   const float* __restrict__ bias,
                                   void* __restrict__ outv, int N, int mode) {
  __shared__ __align__(16) __bf16 sA[16 * 32];
  const int lane = threadIdx.x & 31;
  const int wave = threadIdx.x >> 5;
  const int ntile = blockIdx.x * 4 + wave;
  const int ntiles = (N + 15) >> 4;
  const int Ksteps = (K + 31) >> 5;
  const bool nvalid = (ntile < ntiles);
  v8f acc = {};
  for (int kt = 0; kt < Ksteps; ++kt) {
    for (int i = threadIdx.x; i < 512; i += 128) {
      int row = i >> 5, k = (kt << 5) + (i & 31);
      sA[i] = f2bf(k < K ? A[row * K + k] : 0.f);
    }
    __syncthreads();
    if (nvalid) {
      if (kt + 1 < Ksteps)
        __builtin_prefetch(Bpk + ((long)ntile * Ksteps + kt + 1) * 512, 0, 1);
      v16bf a = lds_a_frag(sA, 32, 0, lane);
      v16bf b = load_b_frag(Bpk, (long)ntile * Ksteps + kt, lane);
      acc = wmma_bf16(a, b, acc);
    }
    __syncthreads();
  }
  if (!nvalid) return;
  int n = (ntile << 4) + (lane & 15);
  if (n >= N) return;
  float bs = bias ? bias[n] : 0.f;
  for (int r = 0; r < 8; ++r) {
    int row = r + ((lane >> 4) << 3);
    float v = acc[r] + bs;
    if (mode == 0) ((float*)outv)[row * N + n] = v;
    else {
      int l = n >> 10, j = n & 1023;
      long o = ((long)(l << 4) + row) * 1024 + j;
      if (mode == 1) ((float*)outv)[o] = v;
      else           ((__bf16*)outv)[o] = f2bf(v);
    }
  }
}

// ---------------------------------------------------------------------------
// Fused LSTM layer step. x/hprev/hcur are bf16 [16][Kx]/[16][1024]; c is f32.
// 64 blocks x 128 thr; block owns hidden cols [j0,j0+16); wave g owns gate g
// (torch i,f,g,o order). x then h are async-copied whole into LDS (32KB max),
// so the WMMA K-loops run with no barriers and no format conversion.
// ---------------------------------------------------------------------------
__global__ void lstm_layer_kernel(const __bf16* __restrict__ x, int Kx,
                                  const __bf16* __restrict__ wih,
                                  const __bf16* __restrict__ whh,
                                  const float* __restrict__ bih,
                                  const float* __restrict__ bhh,
                                  const __bf16* __restrict__ hprev,
                                  __bf16* __restrict__ hcur,
                                  float* __restrict__ cst) {
  __shared__ __align__(16) __bf16 sH[16 * 1024];   // 32KB staging (x, then h)
  __shared__ float gsm[4][16][16];
  const int lane = threadIdx.x & 31;
  const int wave = threadIdx.x >> 5;
  const int j0 = blockIdx.x << 4;
  const int ncol = (wave << 10) + j0;              // gate column base in 4096
  const int ntile = ncol >> 4;
  v8f acc = {};

  // ---- phase 1: x @ W_ih^T ----
  const int ks1 = (Kx + 31) >> 5;
  {
    int nch = (16 * Kx) >> 3;                      // 16B chunks of 16*Kx bf16
    for (int i = threadIdx.x; i < nch; i += 128)
      async_g2l_b128((char*)sH + i * 16, (const char*)x + i * 16);
    wait_async0();
    __syncthreads();
    for (int kt = 0; kt < ks1; ++kt) {
      v16bf a = lds_a_frag(sH, Kx, kt << 5, lane);
      v16bf b = load_b_frag(wih, (long)ntile * ks1 + kt, lane);
      acc = wmma_bf16(a, b, acc);
    }
    __syncthreads();                               // before re-staging sH
  }
  // ---- phase 2: h @ W_hh^T (K=1024; packed W_hh L2-resident all 100 steps) ----
  {
    for (int i = threadIdx.x; i < 2048; i += 128)  // 32KB / 16B
      async_g2l_b128((char*)sH + i * 16, (const char*)hprev + i * 16);
    wait_async0();
    __syncthreads();
    for (int kt = 0; kt < 32; ++kt) {
      if (kt + 1 < 32)
        __builtin_prefetch(whh + ((long)ntile * 32 + kt + 1) * 512, 0, 1);
      v16bf a = lds_a_frag(sH, 1024, kt << 5, lane);
      v16bf b = load_b_frag(whh, (long)ntile * 32 + kt, lane);
      acc = wmma_bf16(a, b, acc);
    }
  }
  int col = ncol + (lane & 15);
  float bsum = bih[col] + bhh[col];
  for (int r = 0; r < 8; ++r) {
    int row = r + ((lane >> 4) << 3);
    gsm[wave][row][lane & 15] = acc[r] + bsum;
  }
  __syncthreads();
  // cell update: 256 (row,col) elements with 128 threads
  for (int it = 0; it < 2; ++it) {
    int idx = it * 128 + threadIdx.x;
    int row = idx >> 4, cc = idx & 15;
    int j = j0 + cc;
    float ig = gsm[0][row][cc], fg = gsm[1][row][cc];
    float gg = gsm[2][row][cc], og = gsm[3][row][cc];
    float cold = cst[row * 1024 + j];
    float cn = sigf(fg) * cold + sigf(ig) * tanhf(gg);
    float hn = sigf(og) * tanhf(cn);
    cst[row * 1024 + j] = cn;
    hcur[row * 1024 + j] = f2bf(hn);
  }
}

// out = sigmoid(h3 @ lin_w^T + lin_b); writes d_out[b][t][o] and next-step x0.
__global__ void out_step_kernel(const __bf16* __restrict__ h3,
                                const float* __restrict__ lw,
                                const float* __restrict__ lb,
                                float* __restrict__ out, __bf16* __restrict__ x0,
                                int t) {
  int tid = threadIdx.x;
  if (tid >= 16) return;
  int b = tid >> 2, o = tid & 3;
  float s = 0.f;
  for (int k = 0; k < 1024; ++k) s += bf2f(h3[b * 1024 + k]) * lw[o * 1024 + k];
  float v = sigf(s + lb[o]);
  out[(b * 100 + t) * 4 + o] = v;
  x0[b * 32 + o] = f2bf(v);
}

// ---------------------------------------------------------------------------
// Host orchestration
// ---------------------------------------------------------------------------
struct BNp { const float *b, *g, *m, *v; };

extern "C" void kernel_launch(void* const* d_in, const int* in_sizes, int n_in,
                              void* d_out, int out_size, void* d_ws, size_t ws_size,
                              hipStream_t stream) {
  (void)in_sizes; (void)n_in; (void)out_size; (void)ws_size;
  // ---- leaf pointers in jax pytree (alphabetical-key) flatten order ----
  int ix = 0;
  auto F = [&]() { return (const float*)d_in[ix++]; };
  auto getBN = [&]() { BNp r; r.b = F(); r.g = F(); r.m = F(); r.v = F(); return r; };
  BNp bn1 = getBN();
  const float* conv1_b = F();
  const float* conv1_w = F();
  const float* fc_b = F();
  const float* fc_w = F();
  struct Blk { BNp bn1, bn2; const float *c1, *c2, *dc; BNp dbn; bool down; };
  Blk blk[4][2];
  for (int li = 0; li < 4; ++li)
    for (int bi = 0; bi < 2; ++bi) {
      Blk& B = blk[li][bi];
      B.down = (bi == 0 && li > 0);
      B.bn1 = getBN(); B.bn2 = getBN(); B.c1 = F(); B.c2 = F(); B.dc = nullptr;
      if (B.down) { B.dbn = getBN(); B.dc = F(); }
    }
  const float* lin_b = F();
  const float* lin_w = F();
  struct LS { const float *bhh, *bih, *whh, *wih; };
  LS ls[4];
  for (int l = 0; l < 4; ++l) { ls[l].bhh = F(); ls[l].bih = F(); ls[l].whh = F(); ls[l].wih = F(); }
  const float* proj_c_b = F(); const float* proj_c_w = F();
  const float* proj_h_b = F(); const float* proj_h_w = F();
  const float* xin = F();

  // ---- workspace carve (~195 MB) ----
  char* wsb = (char*)d_ws;
  size_t off = 0;
  auto carve = [&](size_t bytes) {
    char* p = wsb + off; off = (off + bytes + 255) & ~(size_t)255; return p;
  };
  float* bufA = (float*)carve((size_t)4 * 64 * 210 * 210 * 4);   // stem out
  float* bufs[4];
  for (int i = 0; i < 4; ++i) bufs[i] = (float*)carve((size_t)12 << 20);
  float*  feat  = (float*)carve(16 * 512 * 4);
  float*  featH = (float*)carve(16 * 1024 * 4);
  __bf16* hA = (__bf16*)carve(4 * 16 * 1024 * 2);
  __bf16* hB = (__bf16*)carve(4 * 16 * 1024 * 2);
  float*  cS = (float*)carve(4 * 16 * 1024 * 4);
  __bf16* x0 = (__bf16*)carve(16 * 32 * 2);

  auto pack = [&](const float* src, int N, int K) -> const __bf16* {
    int nt = (N + 15) / 16, ks = (K + 31) / 32;
    long total = (long)nt * ks * 512;
    __bf16* dst = (__bf16*)carve((size_t)total * 2);
    pack_b_kernel<<<(unsigned)((total + 255) / 256), 256, 0, stream>>>(src, dst, N, K, ks, total);
    return dst;
  };

  const int chans[5] = {64, 64, 128, 256, 512};
  const __bf16* stem_pk = pack(conv1_w, 64, 625);
  const __bf16 *c1pk[4][2], *c2pk[4][2], *dcpk[4];
  for (int li = 0; li < 4; ++li)
    for (int bi = 0; bi < 2; ++bi) {
      int cin = (bi == 0) ? chans[li] : chans[li + 1];
      int cout = chans[li + 1];
      c1pk[li][bi] = pack(blk[li][bi].c1, cout, cin * 9);
      c2pk[li][bi] = pack(blk[li][bi].c2, cout, cout * 9);
      if (blk[li][bi].down) dcpk[li] = pack(blk[li][bi].dc, cout, cin);
    }
  const __bf16* fc_pk = pack(fc_w, 1024, 512);
  const __bf16* ph_pk = pack(proj_h_w, 4096, 1024);
  const __bf16* pc_pk = pack(proj_c_w, 4096, 1024);
  const __bf16 *wih_pk[4], *whh_pk[4];
  wih_pk[0] = pack(ls[0].wih, 4096, 4);
  for (int l = 1; l < 4; ++l) wih_pk[l] = pack(ls[l].wih, 4096, 1024);
  for (int l = 0; l < 4; ++l) whh_pk[l] = pack(ls[l].whh, 4096, 1024);

  auto conv = [&](const float* in, const __bf16* wpk, const float* bias, const BNp* bn,
                  const float* resid, float* out, int Bn, int Cin, int H, int W,
                  int Cout, int Ho, int Wo, int KH, int KW, int s, int p, int relu) {
    int mt = (Bn * Ho * Wo + 15) / 16;
    int ng = ((Cout + 15) / 16 + 3) / 4;
    conv_wmma_kernel<<<dim3(mt, ng), 128, 0, stream>>>(
        in, wpk, bias, bn ? bn->g : nullptr, bn ? bn->b : nullptr,
        bn ? bn->m : nullptr, bn ? bn->v : nullptr, resid, out,
        Bn, Cin, H, W, Cout, Ho, Wo, KH, KW, s, p, relu);
  };

  // ---- CNN ----
  conv(xin, stem_pk, conv1_b, &bn1, nullptr, bufA, 4, 1, 224, 224, 64, 210, 210,
       25, 25, 1, 5, 1);
  {
    long total = (long)4 * 64 * 105 * 105;
    maxpool_kernel<<<(unsigned)((total + 255) / 256), 256, 0, stream>>>(
        bufA, bufs[0], 4, 64, 210, 210, 105, 105);
  }
  float* cur = bufs[0];
  int icur = 0, H = 105, W = 105, C = 64;
  for (int li = 0; li < 4; ++li)
    for (int bi = 0; bi < 2; ++bi) {
      Blk& B = blk[li][bi];
      int cout = chans[li + 1];
      int s = (bi == 0 && li > 0) ? 2 : 1;
      int Ho = (H + 2 - 3) / s + 1, Wo = Ho;
      float* t1 = bufs[(icur + 1) & 3];
      float* t2 = bufs[(icur + 2) & 3];
      float* sb = bufs[(icur + 3) & 3];
      conv(cur, c1pk[li][bi], nullptr, &B.bn1, nullptr, t1, 4, C, H, W, cout, Ho, Wo,
           3, 3, s, 1, 1);
      const float* sc = cur;
      if (B.down) {
        conv(cur, dcpk[li], nullptr, &B.dbn, nullptr, sb, 4, C, H, W, cout, Ho, Wo,
             1, 1, s, 0, 0);
        sc = sb;
      }
      conv(t1, c2pk[li][bi], nullptr, &B.bn2, sc, t2, 4, cout, Ho, Wo, cout, Ho, Wo,
           3, 3, 1, 1, 1);
      icur = (icur + 2) & 3; cur = t2; H = Ho; W = Wo; C = cout;
    }
  avgpool_kernel<<<(16 * 512 + 255) / 256, 256, 0, stream>>>(cur, feat, 4, 512, 14 * 14);

  // fc 512->1024 (f32), proj_h -> bf16 h state, proj_c -> f32 c state
  gemm16_wmma_kernel<<<16, 128, 0, stream>>>(feat, 512, fc_pk, fc_b, featH, 1024, 0);
  gemm16_wmma_kernel<<<64, 128, 0, stream>>>(featH, 1024, ph_pk, proj_h_b, hA, 4096, 2);
  gemm16_wmma_kernel<<<64, 128, 0, stream>>>(featH, 1024, pc_pk, proj_c_b, cS, 4096, 1);
  zero_kernel<<<1, 256, 0, stream>>>((float*)x0, 256);   // 512 bf16 zeros

  // ---- LSTM: 100 steps x 4 layers, h ping-pong per step ----
  float* outp = (float*)d_out;
  for (int t = 0; t < 100; ++t) {
    __bf16* hp = (t & 1) ? hB : hA;
    __bf16* hc = (t & 1) ? hA : hB;
    lstm_layer_kernel<<<64, 128, 0, stream>>>(x0, 32, wih_pk[0], whh_pk[0],
                                              ls[0].bih, ls[0].bhh, hp, hc, cS);
    for (int l = 1; l < 4; ++l)
      lstm_layer_kernel<<<64, 128, 0, stream>>>(hc + (l - 1) * 16 * 1024, 1024,
                                                wih_pk[l], whh_pk[l],
                                                ls[l].bih, ls[l].bhh,
                                                hp + l * 16 * 1024,
                                                hc + l * 16 * 1024,
                                                cS + l * 16 * 1024);
    out_step_kernel<<<1, 32, 0, stream>>>(hc + 3 * 16 * 1024, lin_w, lin_b, outp, x0, t);
  }
}
